// HiResidualVQ_11519102288388
// MI455X (gfx1250) — matrix operand order repaired and checked by reading, not compiled, not run on hardware
//
#include <hip/hip_runtime.h>
#include <math.h>

// Problem constants (from reference)
#define NQv   6
#define NBc   1024            // codes per codebook
#define CDv   512             // channel dim
#define Bv    64
#define Tv    384
#define BTv   (Bv * Tv)       // 24576
#define K2v   (2 * CDv)       // 1024 (conv input channels)
#define CTv   ((size_t)CDv * Tv)

typedef __attribute__((ext_vector_type(16))) __bf16 v16bf;
typedef __attribute__((ext_vector_type(8)))  __bf16 v8bf;
typedef __attribute__((ext_vector_type(8)))  float  v8f;

__device__ __forceinline__ v16bf join16(v8bf a, v8bf b) {
  return __builtin_shufflevector(a, b, 0,1,2,3,4,5,6,7,8,9,10,11,12,13,14,15);
}

// ---------------- elementwise / staging kernels ----------------

__global__ void k_zero_f32(float* p, size_t n) {
  size_t i = (size_t)blockIdx.x * blockDim.x + threadIdx.x;
  if (i < n) p[i] = 0.0f;
}

__global__ void k_zero_i32(int* p, int n) {
  int i = blockIdx.x * blockDim.x + threadIdx.x;
  if (i < n) p[i] = 0;
}

__global__ void k_cast_bf16(const float* __restrict__ s, __bf16* __restrict__ d, size_t n) {
  size_t i = (size_t)blockIdx.x * blockDim.x + threadIdx.x;
  if (i < n) d[i] = (__bf16)s[i];
}

// W_conv (O=512, I=1024, 3) -> 3 tap matrices, each (O x I) row-major bf16
__global__ void k_cast_wconv(const float* __restrict__ s, __bf16* __restrict__ d) {
  size_t n = (size_t)CDv * K2v * 3;
  size_t i = (size_t)blockIdx.x * blockDim.x + threadIdx.x;
  if (i >= n) return;
  int kk = (int)(i / ((size_t)CDv * K2v));
  size_t r = i % ((size_t)CDv * K2v);
  int o  = (int)(r / K2v);
  int ii = (int)(r % K2v);
  d[i] = (__bf16)s[(size_t)o * K2v * 3 + (size_t)ii * 3 + kk];
}

// 0.5 * ||code||^2 per code row
__global__ void k_cnorm(const float* __restrict__ cb, float* __restrict__ cn, int ncodes) {
  int j = blockIdx.x * blockDim.x + threadIdx.x;
  if (j >= ncodes) return;
  const float* row = cb + (size_t)j * CDv;
  float s = 0.0f;
  for (int c = 0; c < CDv; ++c) { float v = row[c]; s += v * v; }
  cn[j] = 0.5f * s;
}

// src (B, CDv, T) fp32 -> dst[(b*T+t)*S + off + c] bf16   (K-major staging)
__global__ void k_cast_transpose(const float* __restrict__ s, __bf16* __restrict__ d,
                                 int S, int off) {
  size_t n = (size_t)Bv * CTv;
  size_t i = (size_t)blockIdx.x * blockDim.x + threadIdx.x;
  if (i >= n) return;
  int t = (int)(i % Tv);
  size_t r = i / Tv;
  int c = (int)(r % CDv);
  int b = (int)(r / CDv);
  d[((size_t)b * Tv + t) * S + off + c] = (__bf16)s[i];
}

// ---------------- VQ argmin via WMMA ----------------
// A: (BT x 512) bf16 row-major; CB: (1024 x 512) bf16 row-major; cn: 0.5||c||^2
// One wave owns 16 rows; sweeps 1024 codes, 32 at a time with dual accumulators.
__global__ void __launch_bounds__(128)
k_vq_argmin(const __bf16* __restrict__ A, const __bf16* __restrict__ CB,
            const float* __restrict__ cn, int* __restrict__ idx) {
  int wave = (blockIdx.x * blockDim.x + threadIdx.x) >> 5;
  int lane = threadIdx.x & 31;
  int m0 = wave * 16;
  if (m0 >= BTv) return;
  int rr = lane & 15, hi = lane >> 4;

  // Resident A fragments for all 16 K-chunks (K=512, chunks of 32)
  const __bf16* ab = A + (size_t)(m0 + rr) * CDv;
  v16bf a[16];
#pragma unroll
  for (int kc = 0; kc < 16; ++kc) {
    const __bf16* p = ab + kc * 32 + hi * 16;
    v8bf lo = *(const v8bf*)p;
    v8bf h8 = *(const v8bf*)(p + 8);
    a[kc] = join16(lo, h8);
  }

  float bestv[8]; int besti[8];
#pragma unroll
  for (int v = 0; v < 8; ++v) { bestv[v] = -3.4e38f; besti[v] = 0; }

  for (int nt = 0; nt < NBc / 32; ++nt) {
    int n0 = nt * 32;
    const __bf16* bb0 = CB + (size_t)(n0 + rr) * CDv + hi * 16;
    const __bf16* bb1 = CB + (size_t)(n0 + 16 + rr) * CDv + hi * 16;
    // prefetch next tile's codebook rows into cache while we compute
    if (nt + 1 < NBc / 32)
      __builtin_prefetch(CB + (size_t)(n0 + 32 + rr) * CDv + hi * 16, 0, 3);
    v8f acc0 = {}, acc1 = {};
#pragma unroll
    for (int kc = 0; kc < 16; ++kc) {
      v16bf bm0 = *(const v16bf*)(bb0 + kc * 32);
      v16bf bm1 = *(const v16bf*)(bb1 + kc * 32);
      acc0 = __builtin_amdgcn_wmma_f32_16x16x32_bf16(false, a[kc], false, bm0,
                                                     (short)0, acc0, false, false);
      acc1 = __builtin_amdgcn_wmma_f32_16x16x32_bf16(false, a[kc], false, bm1,
                                                     (short)0, acc1, false, false);
    }
    int na = n0 + rr, nb = n0 + 16 + rr;
    float ca = cn[na], cb2 = cn[nb];
#pragma unroll
    for (int v = 0; v < 8; ++v) {
      float s0 = acc0[v] - ca;    // argmax of (x.c - 0.5||c||^2) == argmin dist
      if (s0 > bestv[v]) { bestv[v] = s0; besti[v] = na; }
      float s1 = acc1[v] - cb2;
      if (s1 > bestv[v]) { bestv[v] = s1; besti[v] = nb; }
    }
  }

  // reduce across the 16 lanes sharing each row (xor stays within half)
#pragma unroll
  for (int v = 0; v < 8; ++v) {
    float bv = bestv[v]; int bi = besti[v];
#pragma unroll
    for (int m = 1; m < 16; m <<= 1) {
      float ov = __shfl_xor(bv, m, 32);
      int   oi = __shfl_xor(bi, m, 32);
      if (ov > bv || (ov == bv && oi < bi)) { bv = ov; bi = oi; }
    }
    if (rr == 0) idx[m0 + v + hi * 8] = bi;   // M = v (lanes 0-15) / v+8 (16-31)
  }
}

// ---------------- VQ apply: gather, residual, qsum, commit loss ----------------
__global__ void k_vq_apply(const float* __restrict__ xsrc, float* __restrict__ resid,
                           const float* __restrict__ cb, const int* __restrict__ idx,
                           float* __restrict__ qsum,     // out + stream*CT, batch stride 3*CT
                           __bf16* __restrict__ qbf,     // (BT x CDv) bf16 for linear
                           float* __restrict__ lossAcc) {
  __shared__ float sm[256];
  size_t n = (size_t)Bv * CTv;
  size_t i = (size_t)blockIdx.x * blockDim.x + threadIdx.x;
  float dd = 0.0f;
  if (i < n) {
    int t = (int)(i % Tv);
    size_t r2 = i / Tv;
    int c = (int)(r2 % CDv);
    int b = (int)(r2 / CDv);
    int k = idx[b * Tv + t];
    float qv = cb[(size_t)k * CDv + c];
    float x  = xsrc[i];
    float rv = resid[i];
    resid[i] = rv - qv;
    float df = x - qv;
    dd = df * df;
    qsum[(size_t)b * (3 * CTv) + (size_t)c * Tv + t] += qv;
    qbf[((size_t)b * Tv + t) * CDv + c] = (__bf16)qv;
  }
  sm[threadIdx.x] = dd;
  __syncthreads();
  for (int s = 128; s > 0; s >>= 1) {
    if ((int)threadIdx.x < s) sm[threadIdx.x] += sm[threadIdx.x + s];
    __syncthreads();
  }
  if (threadIdx.x == 0)
    atomicAdd(lossAcc, sm[0] * (1.0f / ((float)BTv * (float)CDv)));
}

// ---------------- histogram + index output, perplexity ----------------
__global__ void k_hist(const int* __restrict__ idx, int* __restrict__ counts,
                       float* __restrict__ idxOut, int stream_i, int qi) {
  int r = blockIdx.x * blockDim.x + threadIdx.x;
  if (r >= BTv) return;
  int k = idx[r];
  atomicAdd(&counts[k], 1);
  int b = r / Tv, t = r % Tv;
  idxOut[(size_t)b * (3 * Tv * NQv) + (size_t)(stream_i * Tv + t) * NQv + qi] = (float)k;
}

__global__ void k_perp(const int* __restrict__ counts, float* __restrict__ perpAcc) {
  __shared__ float sm[1024];
  int j = threadIdx.x;
  float p = (float)counts[j] * (1.0f / (float)BTv);
  sm[j] = -p * logf(p + 1e-10f);
  __syncthreads();
  for (int s = 512; s > 0; s >>= 1) {
    if (j < s) sm[j] += sm[j + s];
    __syncthreads();
  }
  if (j == 0) atomicAdd(perpAcc, expf(sm[0]));
}

// ---------------- Linear (xf @ W^T + b) -> scatter into conv input staging ----------------
// Each wave computes a 16x32 tile (two N-tiles, dual accumulators, A reused).
__global__ void __launch_bounds__(128)
k_linear(const __bf16* __restrict__ A, const __bf16* __restrict__ W,
         const float* __restrict__ bias, __bf16* __restrict__ convin) {
  int wave = (blockIdx.x * blockDim.x + threadIdx.x) >> 5;
  int lane = threadIdx.x & 31;
  const int ntiles = CDv / 32;        // 16
  int tm = wave / ntiles, tn = wave % ntiles;
  int m0 = tm * 16, n0 = tn * 32;
  if (m0 >= BTv) return;
  int rr = lane & 15, hi = lane >> 4;

  const __bf16* ab  = A + (size_t)(m0 + rr) * CDv;
  const __bf16* wb0 = W + (size_t)(n0 + rr) * CDv + hi * 16;
  const __bf16* wb1 = W + (size_t)(n0 + 16 + rr) * CDv + hi * 16;
  v8f acc0 = {}, acc1 = {};
#pragma unroll
  for (int kc = 0; kc < 16; ++kc) {
    const __bf16* p = ab + kc * 32 + hi * 16;
    v8bf lo = *(const v8bf*)p;
    v8bf h8 = *(const v8bf*)(p + 8);
    v16bf am = join16(lo, h8);
    v16bf bm0 = *(const v16bf*)(wb0 + kc * 32);
    v16bf bm1 = *(const v16bf*)(wb1 + kc * 32);
    acc0 = __builtin_amdgcn_wmma_f32_16x16x32_bf16(false, am, false, bm0,
                                                   (short)0, acc0, false, false);
    acc1 = __builtin_amdgcn_wmma_f32_16x16x32_bf16(false, am, false, bm1,
                                                   (short)0, acc1, false, false);
  }
  int na = n0 + rr, nb = n0 + 16 + rr;
  float bva = bias[na], bvb = bias[nb];
#pragma unroll
  for (int v = 0; v < 8; ++v) {
    // flat index in y (BT x C) row-major == flat (B, C, T) index (the faithful reshape quirk)
    size_t f0 = (size_t)(m0 + v + 8 * hi) * CDv + na;
    int b0 = (int)(f0 / CTv);
    size_t rem0 = f0 % CTv;
    convin[((size_t)b0 * Tv + (int)(rem0 % Tv)) * K2v + (int)(rem0 / Tv)] =
        (__bf16)(acc0[v] + bva);
    size_t f1 = (size_t)(m0 + v + 8 * hi) * CDv + nb;
    int b1 = (int)(f1 / CTv);
    size_t rem1 = f1 % CTv;
    convin[((size_t)b1 * Tv + (int)(rem1 % Tv)) * K2v + (int)(rem1 / Tv)] =
        (__bf16)(acc1[v] + bvb);
  }
}

// ---------------- Conv1d k=3 pad=1 via WMMA ----------------
// inT: (B, T, 2C) bf16; Wt: 3 x (512 x 1024) bf16; out: (B, 512, T) fp32
// Each wave computes 16 output channels x 32 time steps (weights reused across 2 t-tiles).
__global__ void __launch_bounds__(128)
k_conv(const __bf16* __restrict__ inT, const __bf16* __restrict__ Wt,
       const float* __restrict__ bias, float* __restrict__ out) {
  int wave = (blockIdx.x * blockDim.x + threadIdx.x) >> 5;
  int lane = threadIdx.x & 31;
  const int OT = CDv / 16;   // 32
  const int TL = Tv / 32;    // 12
  int b = wave / (OT * TL);
  if (b >= Bv) return;
  int r2 = wave % (OT * TL);
  int o0 = (r2 / TL) * 16, t0 = (r2 % TL) * 32;
  int rr = lane & 15, hi = lane >> 4;

  v8f acc0 = {}, acc1 = {};
  const v16bf zf = {};
  for (int kk = 0; kk < 3; ++kk) {
    int tp0 = t0 + rr + kk - 1;
    int tp1 = t0 + 16 + rr + kk - 1;
    bool ok0 = (tp0 >= 0) && (tp0 < Tv);
    bool ok1 = (tp1 >= 0) && (tp1 < Tv);
    const __bf16* br0 = inT + ((size_t)b * Tv + (ok0 ? tp0 : 0)) * K2v + hi * 16;
    const __bf16* br1 = inT + ((size_t)b * Tv + (ok1 ? tp1 : 0)) * K2v + hi * 16;
    const __bf16* ar  = Wt + ((size_t)kk * CDv + o0 + rr) * K2v + hi * 16;
#pragma unroll 8
    for (int kc = 0; kc < 32; ++kc) {
      const __bf16* p = ar + kc * 32;
      v8bf lo = *(const v8bf*)p;
      v8bf h8 = *(const v8bf*)(p + 8);
      v16bf am = join16(lo, h8);
      v16bf bm0 = *(const v16bf*)(br0 + kc * 32);
      if (!ok0) bm0 = zf;                    // zero-pad boundary columns
      v16bf bm1 = *(const v16bf*)(br1 + kc * 32);
      if (!ok1) bm1 = zf;
      acc0 = __builtin_amdgcn_wmma_f32_16x16x32_bf16(false, am, false, bm0,
                                                     (short)0, acc0, false, false);
      acc1 = __builtin_amdgcn_wmma_f32_16x16x32_bf16(false, am, false, bm1,
                                                     (short)0, acc1, false, false);
    }
  }
#pragma unroll
  for (int v = 0; v < 8; ++v) {
    int o = o0 + v + 8 * hi;
    float bo = bias[o];
    out[(size_t)b * CTv + (size_t)o * Tv + t0 + rr]      = acc0[v] + bo;
    out[(size_t)b * CTv + (size_t)o * Tv + t0 + 16 + rr] = acc1[v] + bo;
  }
}

__global__ void k_final(const float* __restrict__ acc, float* __restrict__ o2) {
  o2[0] = acc[0] * (1.0f / NQv);
  o2[1] = acc[1] * (1.0f / NQv);
}

// ---------------- host orchestration ----------------
extern "C" void kernel_launch(void* const* d_in, const int* in_sizes, int n_in,
                              void* d_out, int out_size, void* d_ws, size_t ws_size,
                              hipStream_t stream) {
  const float* x_body  = (const float*)d_in[0];
  const float* x_hands = (const float*)d_in[1];
  const float* x_face  = (const float*)d_in[2];
  const float* cb_f32[3] = { (const float*)d_in[3], (const float*)d_in[4], (const float*)d_in[5] };
  const float* W_lin  = (const float*)d_in[6];
  const float* b_lin  = (const float*)d_in[7];
  const float* W_conv = (const float*)d_in[8];
  const float* b_conv = (const float*)d_in[9];
  float* out = (float*)d_out;

  char* ws = (char*)d_ws;
  size_t off = 0;
  auto alloc = [&](size_t bytes) -> char* {
    char* p = ws + off;
    off = (off + bytes + 255) & ~(size_t)255;
    return p;
  };
  const size_t xbytes = (size_t)Bv * CTv * sizeof(float);
  float*  rb     = (float*)alloc(xbytes);
  float*  rh     = (float*)alloc(xbytes);
  float*  rf     = (float*)alloc(xbytes);
  __bf16* cbbf[3];
  for (int s = 0; s < 3; ++s) cbbf[s] = (__bf16*)alloc((size_t)NQv * NBc * CDv * 2);
  float*  cn     = (float*)alloc((size_t)3 * NQv * NBc * 4);
  __bf16* wlbf   = (__bf16*)alloc((size_t)CDv * CDv * 2);
  __bf16* wcbf   = (__bf16*)alloc((size_t)3 * CDv * K2v * 2);
  __bf16* xfbf   = (__bf16*)alloc((size_t)BTv * CDv * 2);
  __bf16* qbf    = (__bf16*)alloc((size_t)BTv * CDv * 2);
  __bf16* convin = (__bf16*)alloc((size_t)Bv * Tv * K2v * 2);
  float*  hiera  = (float*)alloc(xbytes);
  int*    idx    = (int*)alloc((size_t)BTv * 4);
  int*    counts = (int*)alloc((size_t)NBc * 4);
  float*  accum  = (float*)alloc(256);   // [0]=loss, [1]=perp

  const size_t Q0 = (size_t)Bv * 3 * CTv;                 // quantized_out elems
  const size_t I0 = (size_t)Bv * 3 * Tv * NQv;            // indices elems
  const size_t nel = (size_t)Bv * CTv;

  auto g = [](size_t n, int blk) { return (unsigned)((n + (size_t)blk - 1) / blk); };

  // residual working copies
  hipMemcpyAsync(rb, x_body,  xbytes, hipMemcpyDeviceToDevice, stream);
  hipMemcpyAsync(rh, x_hands, xbytes, hipMemcpyDeviceToDevice, stream);
  hipMemcpyAsync(rf, x_face,  xbytes, hipMemcpyDeviceToDevice, stream);

  // zero qsum region of out + accumulators
  k_zero_f32<<<g(Q0, 256), 256, 0, stream>>>(out, Q0);
  k_zero_f32<<<1, 2, 0, stream>>>(accum, 2);

  // stage codebooks / weights to bf16, code norms
  const size_t cbn = (size_t)NQv * NBc * CDv;
  for (int s = 0; s < 3; ++s) {
    k_cast_bf16<<<g(cbn, 256), 256, 0, stream>>>(cb_f32[s], cbbf[s], cbn);
    k_cnorm<<<g(NQv * NBc, 256), 256, 0, stream>>>(cb_f32[s], cn + (size_t)s * NQv * NBc, NQv * NBc);
  }
  k_cast_bf16<<<g((size_t)CDv * CDv, 256), 256, 0, stream>>>(W_lin, wlbf, (size_t)CDv * CDv);
  k_cast_wconv<<<g((size_t)3 * CDv * K2v, 256), 256, 0, stream>>>(W_conv, wcbf);

  const unsigned vqBlocks   = (BTv / 16) / 4;                    // 4 waves / 128-thread block
  const unsigned gemmBlocks = ((BTv / 16) * (CDv / 32)) / 4;     // 16x32 tiles
  const unsigned convBlocks = (Bv * (CDv / 16) * (Tv / 32)) / 4; // 16x32 tiles

  for (int q = 0; q < NQv; ++q) {
    float* resid_of[3] = { rb, rh, rf };
    for (int s = 0; s < 3; ++s) {
      const float*  cbq   = cb_f32[s] + (size_t)q * NBc * CDv;
      const __bf16* cbqbf = cbbf[s]   + (size_t)q * NBc * CDv;
      const float*  cnq   = cn + ((size_t)s * NQv + q) * NBc;
      const float*  xsrc;
      if (s == 0) {
        xsrc = rb;  // VQ directly on the body residual
      } else {
        // hiera = conv(cat([linear(prev quant), resid_stream], ch), W_conv) per reference
        k_linear<<<gemmBlocks, 128, 0, stream>>>(qbf, wlbf, b_lin, convin);
        k_cast_transpose<<<g(nel, 256), 256, 0, stream>>>(resid_of[s], convin, K2v, CDv);
        k_conv<<<convBlocks, 128, 0, stream>>>(convin, wcbf, b_conv, hiera);
        xsrc = hiera;
      }
      // stage VQ input as (BT x C) bf16, find nearest codes, apply
      k_cast_transpose<<<g(nel, 256), 256, 0, stream>>>(xsrc, xfbf, CDv, 0);
      k_vq_argmin<<<vqBlocks, 128, 0, stream>>>(xfbf, cbqbf, cnq, idx);
      k_vq_apply<<<g(nel, 256), 256, 0, stream>>>(xsrc, resid_of[s], cbq, idx,
                                                  out + (size_t)s * CTv, qbf, accum);
      // histogram -> perplexity, and write indices (as float) into d_out
      k_zero_i32<<<g(NBc, 256), 256, 0, stream>>>(counts, NBc);
      k_hist<<<g(BTv, 256), 256, 0, stream>>>(idx, counts, out + Q0, s, q);
      k_perp<<<1, 1024, 0, stream>>>(counts, accum + 1);
    }
  }

  k_final<<<1, 1, 0, stream>>>(accum, out + Q0 + I0);
}
